// TMTLayer_82471962018132
// MI455X (gfx1250) — matrix operand order repaired and checked by reading, not compiled, not run on hardware
//
#include <hip/hip_runtime.h>

// ---------------------------------------------------------------------------
// MI455X / gfx1250, wave32. Matrix math via v_wmma_f32_16x16x32_bf16 (f32
// accumulate). B-panel staging via async global->LDS copies (ctx) and the
// Tensor Data Mover (mlp1). Inversion / exp / layernorm stats kept in f32.
// ---------------------------------------------------------------------------

typedef __attribute__((ext_vector_type(16))) __bf16 v16bf;
typedef __attribute__((ext_vector_type(8)))  float  v8f;
typedef __attribute__((ext_vector_type(4)))  unsigned v4u;
typedef __attribute__((ext_vector_type(8)))  int    v8i_t;
typedef __attribute__((ext_vector_type(4)))  int    v4i_t;

#define DEV static __device__ __forceinline__

constexpr int BS = 16;     // batch
constexpr int NN = 1024;   // sequence
constexpr int DD = 256;    // feature
constexpr int NITER = 4;

// ---- f32 -> bf16, round-to-nearest-even, no reliance on __bf16 arithmetic --
DEV __bf16 f2bf(float f) {
  unsigned u = __builtin_bit_cast(unsigned, f);
  unsigned r = (u + 0x7FFFu + ((u >> 16) & 1u)) >> 16;
  unsigned short s = (unsigned short)r;
  return __builtin_bit_cast(__bf16, s);
}

// ---- WMMA fragment loaders (contiguous k = 16*half + s, wave32) --------------------
// A (16x32 bf16): lane<16 -> row row0+lane, K = k0+0..15
//                 lane>=16 -> row row0+lane-16, K = k0+16..31
DEV v16bf load_a_frag(const __bf16* A, int lda, int row0, int k0) {
  int lane = threadIdx.x & 31;
  int half = lane >> 4;
  const __bf16* p = A + (size_t)(row0 + (lane & 15)) * lda + k0 + half * 16;
  v16bf out;
#pragma unroll
  for (int i = 0; i < 8; ++i) out[i] = p[i];
#pragma unroll
  for (int i = 0; i < 8; ++i) out[8 + i] = p[8 + i];
  return out;
}

// B (32x16 bf16), Bcm stored column-major as rows of length ldb (row = out col):
// lane<16 -> col col0+lane, K = k0+0..15 ; lane>=16 -> col, K = k0+16..31
DEV v16bf load_b_frag(const __bf16* Bcm, int ldb, int col0, int k0) {
  int lane = threadIdx.x & 31;
  int half = lane >> 4;
  const __bf16* p = Bcm + (size_t)(col0 + (lane & 15)) * ldb + k0 + half * 16;
  v16bf out;
#pragma unroll
  for (int i = 0; i < 16; ++i) out[i] = p[i];
  return out;
}

DEV v8f wmma_bf16(v16bf a, v16bf b, v8f c) {
  return __builtin_amdgcn_wmma_f32_16x16x32_bf16(false, a, false, b,
                                                 (short)0, c, false, false);
}

DEV v8f gemm_tile(const __bf16* A, int lda, const __bf16* Bcm, int ldb,
                  int row0, int col0, int K) {
  v8f acc = {};
  for (int k0 = 0; k0 < K; k0 += 32) {
    v16bf a = load_a_frag(A, lda, row0, k0);
    v16bf b = load_b_frag(Bcm, ldb, col0, k0);
    acc = wmma_bf16(a, b, acc);
  }
  return acc;
}

// C/D layout: VGPR i -> M = row0 + i (lanes 0..15) or row0 + 8 + i (lanes 16..31),
// N = col0 + (lane & 15).
DEV void cd_coords(int row0, int col0, int& mbase, int& n) {
  int lane = threadIdx.x & 31;
  n = col0 + (lane & 15);
  mbase = row0 + ((lane >> 4) ? 8 : 0);
}

// ---------------------------------------------------------------------------
// Elementwise / prep kernels
// ---------------------------------------------------------------------------
__global__ void f32_to_bf16_kernel(const float* __restrict__ s, __bf16* __restrict__ d, long n) {
  long i = (long)blockIdx.x * blockDim.x + threadIdx.x;
  if (i < n) d[i] = f2bf(s[i]);
}

// src (nmat, rows, cols) f32 -> dst (nmat, cols, rows) bf16
__global__ void transpose_f32_to_bf16(const float* __restrict__ src, __bf16* __restrict__ dst,
                                      int rows, int cols, long total) {
  long idx = (long)blockIdx.x * blockDim.x + threadIdx.x;
  if (idx >= total) return;
  long per = (long)rows * cols;
  long m = idx / per;
  long rem = idx - m * per;
  int c = (int)(rem / rows);
  int r = (int)(rem - (long)c * rows);
  dst[idx] = f2bf(src[(m * rows + r) * (long)cols + c]);
}

// x (B*N, D) f32 -> low half of h (B*N, 2D) bf16
__global__ void pack_x_to_h(const float* __restrict__ x, __bf16* __restrict__ h) {
  long i = (long)blockIdx.x * blockDim.x + threadIdx.x;
  long row = i >> 8;
  int  c = (int)(i & 255);
  h[row * (2 * DD) + c] = f2bf(x[i]);
}

// root = exp(max(sv.Wr + br - mask*50, -40)) ; one block per (b,n) row
__global__ void root_kernel(const float* __restrict__ sv, const float* __restrict__ Wr,
                            const float* __restrict__ br, const float* __restrict__ mask,
                            float* __restrict__ Rexp) {
  int row = blockIdx.x;
  int d = threadIdx.x;
  __shared__ float sh[DD];
  sh[d] = sv[(size_t)row * DD + d] * Wr[d];
  __syncthreads();
  for (int s = DD / 2; s > 0; s >>= 1) {
    if (d < s) sh[d] += sh[d + s];
    __syncthreads();
  }
  if (d == 0) {
    float r0 = sh[0] + br[0] - mask[row] * 50.f;
    Rexp[row] = __expf(fmaxf(r0, -40.f));
  }
}

// L[b,j] = sum_i A[b,i,j]
__global__ void colsum_kernel(const float* __restrict__ Amat, float* __restrict__ Lsum) {
  int idx = blockIdx.x * blockDim.x + threadIdx.x;   // b*N + j
  int b = idx >> 10, j = idx & (NN - 1);
  const float* Ab = Amat + (size_t)b * NN * NN;
  float s = 0.f;
  for (int i = 0; i < NN; ++i) s += Ab[(size_t)i * NN + j];
  Lsum[idx] = s;
}

// LL[b,i,j] = -A[b,i,j] + (i==j) * (L[b,i] + R[b,i])
__global__ void build_LL(const float* __restrict__ Amat, const float* __restrict__ Lsum,
                         const float* __restrict__ Rexp, float* __restrict__ LL) {
  size_t idx = (size_t)blockIdx.x * blockDim.x + threadIdx.x;
  size_t b = idx >> 20;
  size_t rem = idx & ((size_t)NN * NN - 1);
  int i = (int)(rem >> 10), j = (int)(rem & (NN - 1));
  float v = -Amat[idx];
  if (i == j) v += Lsum[b * NN + i] + Rexp[b * NN + i];
  LL[idx] = v;
}

// In-place Gauss-Jordan inverse (no pivoting; LL is diagonally dominant).
// One 1024-thread workgroup per batch matrix; thread owns one column.
__global__ void gj_inverse(float* __restrict__ M) {
  int b = blockIdx.x;
  float* Mb = M + (size_t)b * NN * NN;
  __shared__ float colp[NN];
  __shared__ float s_invp;
  int c = threadIdx.x;
  for (int p = 0; p < NN; ++p) {
    colp[c] = Mb[(size_t)c * NN + p];
    if (c == 0) s_invp = 1.f / Mb[(size_t)p * NN + p];
    __syncthreads();
    float invp = s_invp;
    float Mpc = Mb[(size_t)p * NN + c];
    float newMpc = (c == p) ? invp : Mpc * invp;
    Mb[(size_t)p * NN + c] = newMpc;
    for (int r = 0; r < NN; ++r) {
      if (r == p) continue;
      float f = colp[r];
      float cur = Mb[(size_t)r * NN + c];
      float val = (c == p) ? (-f * invp) : (cur - f * newMpc);
      Mb[(size_t)r * NN + c] = val;
    }
    __syncthreads();
  }
}

// attn[b,n,m] = mask[b,m] ? 0 : A[b,m,n] * (LLinv[b,n,n] - LLinv[b,n,m])
// d0[b,n] = R[b,n] * LLinv[b,n,n]
__global__ void attn_kernel(const float* __restrict__ Amat, const float* __restrict__ LLinv,
                            const float* __restrict__ mask, const float* __restrict__ Rexp,
                            float* __restrict__ out_attn, __bf16* __restrict__ attnb,
                            float* __restrict__ out_d0) {
  int m = blockIdx.x * blockDim.x + threadIdx.x;
  int n = blockIdx.y;
  int b = blockIdx.z;
  const float* Ab = Amat + (size_t)b * NN * NN;
  const float* Lb = LLinv + (size_t)b * NN * NN;
  float diag = Lb[(size_t)n * NN + n];
  float a = (mask[b * NN + m] != 0.f)
                ? 0.f
                : Ab[(size_t)m * NN + n] * (diag - Lb[(size_t)n * NN + m]);
  size_t oi = ((size_t)b * NN + n) * NN + m;
  out_attn[oi] = a;
  attnb[oi] = f2bf(a);
  if (m == n) out_d0[b * NN + n] = Rexp[b * NN + n] * diag;
}

// Layernorm over D; writes bf16 v into h[:, :D], bf16 v^T, and (last iter) f32 out.
__global__ void ln_kernel(const float* __restrict__ h2, const float* __restrict__ g,
                          const float* __restrict__ bta, __bf16* __restrict__ h,
                          __bf16* __restrict__ vT, float* __restrict__ out_v, int write_f32) {
  int row = blockIdx.x;      // b*N + n
  int d = threadIdx.x;       // 0..255
  __shared__ float s1[DD], s2[DD];
  float x = h2[(size_t)row * DD + d];
  s1[d] = x;
  s2[d] = x * x;
  __syncthreads();
  for (int s = DD / 2; s > 0; s >>= 1) {
    if (d < s) { s1[d] += s1[d + s]; s2[d] += s2[d + s]; }
    __syncthreads();
  }
  float mu = s1[0] * (1.f / DD);
  float var = s2[0] * (1.f / DD) - mu * mu;
  float y = (x - mu) * rsqrtf(var + 1e-6f) * g[d] + bta[d];
  h[(size_t)row * (2 * DD) + d] = f2bf(y);
  int b = row >> 10, n = row & (NN - 1);
  vT[((size_t)b * DD + d) * NN + n] = f2bf(y);
  if (write_f32) out_v[(size_t)row * DD + d] = y;
}

// ---------------------------------------------------------------------------
// WMMA GEMM kernels: 256 threads = 8 waves, one 16x16 output tile per wave.
// Grids exactly cover the problem (no divergence at WMMA; EXEC all 1s).
// ---------------------------------------------------------------------------

// out(B*N, D) bf16 = bf16( (sv(B*N,D) @ W + bias) * scale ), W given column-major.
__global__ void proj_gemm(const __bf16* __restrict__ sv, const __bf16* __restrict__ Wt,
                          const float* __restrict__ bias, __bf16* __restrict__ out, float scale) {
  int tile = blockIdx.x * 8 + (threadIdx.x >> 5);    // 16384 tiles: 1024 x 16
  int mt = tile >> 4, nt = tile & 15;
  v8f acc = gemm_tile(sv, DD, Wt, DD, mt * 16, nt * 16, DD);
  int mbase, n; cd_coords(mt * 16, nt * 16, mbase, n);
  float bz = bias[n];
#pragma unroll
  for (int i = 0; i < 8; ++i)
    out[(size_t)(mbase + i) * DD + n] = f2bf((acc[i] + bz) * scale);
}

// A[b] = exp(clamp(q kT - 50*(mask_i + mask_j), -40))  (f32 out)
__global__ void scores_gemm(const __bf16* __restrict__ q, const __bf16* __restrict__ k,
                            const float* __restrict__ mask, float* __restrict__ Amat) {
  int tile = blockIdx.x * 8 + (threadIdx.x >> 5);    // 65536 tiles: 16 x 64 x 64
  int b = tile >> 12;
  int t = tile & 4095;
  int mt = t >> 6, nt = t & 63;
  const __bf16* qb = q + (size_t)b * NN * DD;
  const __bf16* kb = k + (size_t)b * NN * DD;
  v8f acc = gemm_tile(qb, DD, kb, DD, mt * 16, nt * 16, DD);
  int mbase, j; cd_coords(mt * 16, nt * 16, mbase, j);
  float mj = mask[b * NN + j] * 50.f;
#pragma unroll
  for (int i = 0; i < 8; ++i) {
    float sc = acc[i] - mj - mask[b * NN + mbase + i] * 50.f;
    sc = fmaxf(sc, -40.f);
    Amat[((size_t)b * NN + mbase + i) * NN + j] = __expf(sc);
  }
}

// ctx[b](N,D) = attn[b](N,N) @ v[b](N,D). All 8 waves of a block share one
// 16-column B panel (vT rows nt*16..nt*16+15, 16 x 1024 bf16 = 32 KB) which is
// staged whole into LDS with async global->LDS b128 copies.
__global__ void ctx_gemm(const __bf16* __restrict__ attn, const __bf16* __restrict__ vT,
                         __bf16* __restrict__ h) {
  __shared__ __bf16 Bs[16 * NN];                 // 32 KB
  int b   = blockIdx.x >> 7;                     // batch
  int nt  = (blockIdx.x >> 3) & 15;              // column tile (shared by block)
  int mtg = blockIdx.x & 7;                      // row-tile group
  const __bf16* Bg = vT + (size_t)b * DD * NN + (size_t)(nt * 16) * NN; // contiguous 32 KB

  unsigned ldsBase = (unsigned)(size_t)(&Bs[0]);
  {
    int t = threadIdx.x;
#pragma unroll
    for (int i = 0; i < 8; ++i) {
      int e16 = i * 256 + t;                     // 16-byte chunk id, 2048 total
      unsigned lofs = ldsBase + (unsigned)e16 * 16u;
      unsigned long long ga = (unsigned long long)(size_t)(Bg + e16 * 8);
      asm volatile("global_load_async_to_lds_b128 %0, %1, off"
                   :: "v"(lofs), "v"(ga) : "memory");
    }
    asm volatile("s_wait_asynccnt 0x0" ::: "memory");
  }
  __syncthreads();

  int mt = mtg * 8 + (threadIdx.x >> 5);         // 0..63 within batch
  const __bf16* Ab = attn + (size_t)b * NN * NN;
  v8f acc = {};
  for (int k0 = 0; k0 < NN; k0 += 32) {
    if (k0 + 64 < NN)
      __builtin_prefetch(Ab + (size_t)(mt * 16 + (threadIdx.x & 15)) * NN + k0 + 64, 0, 1);
    v16bf a = load_a_frag(Ab, NN, mt * 16, k0);
    v16bf bb = load_b_frag(Bs, NN, 0, k0);       // from LDS
    acc = wmma_bf16(a, bb, acc);
  }
  int mbase, n; cd_coords(mt * 16, nt * 16, mbase, n);
#pragma unroll
  for (int i = 0; i < 8; ++i)
    h[((size_t)b * NN + mbase + i) * (2 * DD) + DD + n] = f2bf(acc[i]);
}

// h1 = relu(h(B*N,2D) @ W1 + b1). The 16-column W1^T panel (16 x 512 bf16 =
// 16 KB) is brought into LDS with a single Tensor Data Mover descriptor.
__global__ void mlp1_gemm(const __bf16* __restrict__ h, const __bf16* __restrict__ W1t,
                          const float* __restrict__ b1, __bf16* __restrict__ h1) {
  __shared__ __bf16 Ws[16 * 2 * DD];             // 16 KB
  int nt  = blockIdx.x & 15;                     // column tile (shared by block)
  int mtg = blockIdx.x >> 4;                     // 0..127
  const __bf16* Bg = W1t + (size_t)(nt * 16) * (2 * DD);

  if (threadIdx.x < 32) {                        // wave 0 owns the TDM op
    unsigned lds = (unsigned)(size_t)(&Ws[0]);
    unsigned long long ga = (unsigned long long)(size_t)Bg;
    v4u g0;
    g0[0] = 1u;                                  // count=1, user mode
    g0[1] = lds;                                 // lds_addr
    g0[2] = (unsigned)(ga & 0xFFFFFFFFu);        // global_addr[31:0]
    g0[3] = (unsigned)((ga >> 32) & 0x1FFFFFFu)  // global_addr[56:32]
            | (2u << 30);                        // type=2 (image)
    v8i_t g1;
    g1[0] = (int)(1u << 16);                     // data_size=1 (2 bytes)
    g1[1] = (int)(512u << 16);                   // tensor_dim0 = 512 (lo16 @ [31:16])
    g1[2] = (int)(16u << 16);                    // dim0 hi=0 | tensor_dim1 = 16
    g1[3] = (int)(512u << 16);                   // dim1 hi=0 | tile_dim0 = 512
    g1[4] = 16;                                  // tile_dim1 = 16, tile_dim2 = 0
    g1[5] = 512;                                 // tensor_dim0_stride = 512 (lo32)
    g1[6] = 0;                                   // stride0 hi | stride1 lo
    g1[7] = 0;
    v4i_t z4 = {};
    v8i_t z8 = {};
    __builtin_amdgcn_tensor_load_to_lds(g0, g1, z4, z4, z8, 0);  // 6-arg form
    __builtin_amdgcn_s_wait_tensorcnt(0);
  }
  __syncthreads();

  int mt = mtg * 8 + (threadIdx.x >> 5);         // 0..1023 over B*N rows
  v8f acc = {};
  for (int k0 = 0; k0 < 2 * DD; k0 += 32) {
    v16bf a = load_a_frag(h, 2 * DD, mt * 16, k0);
    v16bf bb = load_b_frag(Ws, 2 * DD, 0, k0);   // from LDS
    acc = wmma_bf16(a, bb, acc);
  }
  int mbase, n; cd_coords(mt * 16, nt * 16, mbase, n);
  float bz = b1[n];
#pragma unroll
  for (int i = 0; i < 8; ++i)
    h1[(size_t)(mbase + i) * DD + n] = f2bf(fmaxf(acc[i] + bz, 0.f));
}

// h2 = h1(B*N,D) @ W2 + b2, W2 column-major (D, D). f32 out.
__global__ void mlp2_gemm(const __bf16* __restrict__ h1, const __bf16* __restrict__ W2t,
                          const float* __restrict__ b2, float* __restrict__ h2) {
  int tile = blockIdx.x * 8 + (threadIdx.x >> 5);    // 16384 tiles
  int mt = tile >> 4, nt = tile & 15;
  v8f acc = gemm_tile(h1, DD, W2t, DD, mt * 16, nt * 16, DD);
  int mbase, n; cd_coords(mt * 16, nt * 16, mbase, n);
  float bz = b2[n];
#pragma unroll
  for (int i = 0; i < 8; ++i)
    h2[(size_t)(mbase + i) * DD + n] = acc[i] + bz;
}

// ---------------------------------------------------------------------------
extern "C" void kernel_launch(void* const* d_in, const int* in_sizes, int n_in,
                              void* d_out, int out_size, void* d_ws, size_t ws_size,
                              hipStream_t stream) {
  (void)in_sizes; (void)n_in; (void)out_size; (void)ws_size;

  const float* x    = (const float*)d_in[0];
  const float* sv   = (const float*)d_in[1];
  const float* mask = (const float*)d_in[2];
  const float* Wk   = (const float*)d_in[3];
  const float* bk   = (const float*)d_in[4];
  const float* Wq   = (const float*)d_in[5];
  const float* bq   = (const float*)d_in[6];
  const float* Wr   = (const float*)d_in[7];
  const float* br   = (const float*)d_in[8];
  const float* W1   = (const float*)d_in[9];
  const float* b1   = (const float*)d_in[10];
  const float* W2   = (const float*)d_in[11];
  const float* b2   = (const float*)d_in[12];
  const float* lng  = (const float*)d_in[13];
  const float* lnb  = (const float*)d_in[14];

  float* out_v    = (float*)d_out;
  float* out_d0   = out_v + (size_t)BS * NN * DD;
  float* out_attn = out_d0 + (size_t)BS * NN;

  char* w = (char*)d_ws;
  size_t off = 0;
  auto alloc = [&](size_t bytes) -> void* {
    void* p = w + off;
    off = (off + bytes + 255) & ~(size_t)255;
    return p;
  };

  const size_t ND = (size_t)BS * NN * DD;   // 4,194,304
  const size_t N2 = (size_t)BS * NN * NN;   // 16,777,216

  __bf16* svb   = (__bf16*)alloc(ND * 2);
  __bf16* kb    = (__bf16*)alloc(ND * 2);
  __bf16* qb    = (__bf16*)alloc(ND * 2);
  __bf16* WkT   = (__bf16*)alloc((size_t)DD * DD * 2);
  __bf16* WqT   = (__bf16*)alloc((size_t)DD * DD * 2);
  __bf16* W1T   = (__bf16*)alloc((size_t)NITER * DD * 2 * DD * 2);
  __bf16* W2T   = (__bf16*)alloc((size_t)NITER * DD * DD * 2);
  float*  Rexp  = (float*)alloc((size_t)BS * NN * 4);
  float*  Lsum  = (float*)alloc((size_t)BS * NN * 4);
  float*  Amat  = (float*)alloc(N2 * 4);
  float*  LL    = (float*)alloc(N2 * 4);
  __bf16* attnb = (__bf16*)alloc(N2 * 2);
  __bf16* h     = (__bf16*)alloc(ND * 2 * 2);   // (B*N, 2D)
  __bf16* h1    = (__bf16*)alloc(ND * 2);
  float*  h2    = (float*)alloc(ND * 4);
  __bf16* vT    = (__bf16*)alloc(ND * 2);

  // ---- prep: bf16 conversions / weight transposes --------------------------
  f32_to_bf16_kernel<<<(int)(ND / 256), 256, 0, stream>>>(sv, svb, (long)ND);
  transpose_f32_to_bf16<<<(DD * DD + 255) / 256, 256, 0, stream>>>(Wk, WkT, DD, DD, (long)DD * DD);
  transpose_f32_to_bf16<<<(DD * DD + 255) / 256, 256, 0, stream>>>(Wq, WqT, DD, DD, (long)DD * DD);
  transpose_f32_to_bf16<<<(NITER * 2 * DD * DD + 255) / 256, 256, 0, stream>>>(
      W1, W1T, 2 * DD, DD, (long)NITER * 2 * DD * DD);
  transpose_f32_to_bf16<<<(NITER * DD * DD + 255) / 256, 256, 0, stream>>>(
      W2, W2T, DD, DD, (long)NITER * DD * DD);

  // ---- projections (WMMA) --------------------------------------------------
  proj_gemm<<<2048, 256, 0, stream>>>(svb, WkT, bk, kb, 1.0f);
  proj_gemm<<<2048, 256, 0, stream>>>(svb, WqT, bq, qb, 1.0f / 16.0f);  // 1/sqrt(256)
  root_kernel<<<BS * NN, DD, 0, stream>>>(sv, Wr, br, mask, Rexp);

  // ---- scores -> A = exp(...) (WMMA) --------------------------------------
  scores_gemm<<<8192, 256, 0, stream>>>(qb, kb, mask, Amat);
  colsum_kernel<<<BS * NN / 256, 256, 0, stream>>>(Amat, Lsum);
  build_LL<<<(int)(N2 / 256), 256, 0, stream>>>(Amat, Lsum, Rexp, LL);

  // ---- batched Laplacian inverse (f32, in place) ---------------------------
  gj_inverse<<<BS, NN, 0, stream>>>(LL);

  // ---- attn / d0 -----------------------------------------------------------
  attn_kernel<<<dim3(NN / 256, NN, BS), 256, 0, stream>>>(Amat, LL, mask, Rexp,
                                                          out_attn, attnb, out_d0);

  // ---- v0 = x --------------------------------------------------------------
  pack_x_to_h<<<(int)(ND / 256), 256, 0, stream>>>(x, h);
  transpose_f32_to_bf16<<<(int)(ND / 256), 256, 0, stream>>>(x, vT, NN, DD, (long)ND);

  // ---- 4 refinement iterations (WMMA GEMMs + layernorm) --------------------
  for (int it = 0; it < NITER; ++it) {
    ctx_gemm<<<2048, 256, 0, stream>>>(attnb, vT, h);     // async-LDS staged B
    mlp1_gemm<<<2048, 256, 0, stream>>>(h, W1T + (size_t)it * 2 * DD * DD, b1 + it * DD, h1); // TDM-staged W
    mlp2_gemm<<<2048, 256, 0, stream>>>(h1, W2T + (size_t)it * DD * DD, b2 + it * DD, h2);
    ln_kernel<<<BS * NN, DD, 0, stream>>>(h2, lng + it * DD, lnb + it * DD,
                                          h, vT, out_v, it == NITER - 1 ? 1 : 0);
  }
}
